// MoELayer_28681791602837
// MI455X (gfx1250) — matrix operand order repaired and checked by reading, not compiled, not run on hardware
//
#include <hip/hip_runtime.h>
#include <hip/hip_bf16.h>
#include <math.h>

// ---------------- problem constants (match reference) ----------------
#define T_TOKENS 8192          // B*N = 4*2048
#define DIM      1024
#define INNER    4096
#define NEXP     8
#define BM       64            // token tile per ffn block
#define NCHUNK   (INNER / 64)  // 64 inner chunks

// padded LDS leading dims (bf16 halves): 16B-aligned rows, no 16-way bank
// conflicts on fragment loads. LD1/LD2 are produced by TDM pad_enable.
#define LDX 1032   // x tile  [64][1024]  row stride 2064B = 129*16
#define LD1 136    // W1^T    [64][128]   row stride 272B  = 17*16  (128 + 4DW pad)
#define LDH 72     // h tile  [64][64]    row stride 144B  = 9*16
#define LD2 72     // W2^T    [256][64]   (64 + 4DW pad)

typedef __bf16 v16bf __attribute__((ext_vector_type(16)));
typedef float  v8f   __attribute__((ext_vector_type(8)));
typedef unsigned int u32x4 __attribute__((ext_vector_type(4)));
typedef int          i32x8 __attribute__((ext_vector_type(8)));
typedef int          i32x4 __attribute__((ext_vector_type(4)));

union Frag16 { v16bf v; uint4 q[2]; };

__device__ __forceinline__ unsigned short f32_to_bf16_rne(float f) {
    unsigned int u = __float_as_uint(f);
    unsigned int r = u + 0x7FFFu + ((u >> 16) & 1u);   // round-to-nearest-even
    return (unsigned short)(r >> 16);
}

// A fragment (16-bit, 16x32): lanes 0-15: M=lane, halves 0-15 => K+0..15;
// lanes 16-31 add +16 to K base (caller resolves).
__device__ __forceinline__ v16bf lds_load_a(const unsigned short* t, int ld,
                                            int row, int kb) {
    const uint4* p = (const uint4*)(t + row * ld + kb);
    Frag16 f;
    f.q[0] = p[0];
    f.q[1] = p[1];
    return f.v;
}

// B fragment (16-bit, 32x16) from an N-major LDS tile: lane reads 16
// contiguous K halves of row N = n0 + lane%16 (K base includes lane>>4 * 16).
__device__ __forceinline__ v16bf lds_load_b(const unsigned short* t, int ld,
                                            int nrow, int kb) {
    const uint4* p = (const uint4*)(t + nrow * ld + kb);
    Frag16 f;
    f.q[0] = p[0];
    f.q[1] = p[1];
    return f.v;
}

// ---------------- Tensor Data Mover (gfx1250) ----------------
#if defined(__gfx1250__) && __has_builtin(__builtin_amdgcn_tensor_load_to_lds) \
    && __has_builtin(__builtin_amdgcn_s_wait_tensorcnt)
#define USE_TDM 1
#else
#define USE_TDM 0
#endif

#if USE_TDM
// 2D tile load: tile_d1 rows of tile_d0 contiguous bf16 elements, row stride
// stride_d0 elements; LDS row padding via D# pad fields (6-arg clang-23 form;
// groups 2/3 zero: 2D tensor).
__device__ __forceinline__ void tdm_load_2d(unsigned lds_off, const void* gaddr,
                                            unsigned tensor_d0, unsigned tensor_d1,
                                            unsigned stride_d0,
                                            unsigned tile_d0, unsigned tile_d1,
                                            unsigned pad_int_code,
                                            unsigned pad_amt_code) {
    unsigned long long ga = (unsigned long long)(uintptr_t)gaddr;
    u32x4 g0;
    g0[0] = 1u;                                          // count=1, user mode
    g0[1] = lds_off;                                     // lds_addr
    g0[2] = (unsigned)(ga & 0xFFFFFFFFu);                // global_addr[31:0]
    g0[3] = (unsigned)((ga >> 32) & 0x1FFFFFFu) | (2u << 30);  // [56:32] | type=2
    i32x8 g1;
    g1[0] = (int)((1u << 16)                             // data_size = 2 bytes
                  | (1u << 20)                           // pad_enable
                  | (pad_int_code << 22) | (pad_amt_code << 25));
    g1[1] = (int)((tensor_d0 & 0xFFFFu) << 16);          // dim0[15:0] @ bits 63:48
    g1[2] = (int)(((tensor_d0 >> 16) & 0xFFFFu) | ((tensor_d1 & 0xFFFFu) << 16));
    g1[3] = (int)(((tensor_d1 >> 16) & 0xFFFFu) | ((tile_d0 & 0xFFFFu) << 16));
    g1[4] = (int)(tile_d1 & 0xFFFFu);                    // tile_dim1; tile_dim2=0
    g1[5] = (int)stride_d0;                              // dim0_stride[31:0]
    g1[6] = 0;
    g1[7] = 0;
    i32x4 z4 = {0, 0, 0, 0};
    i32x8 z8 = {0, 0, 0, 0, 0, 0, 0, 0};
    __builtin_amdgcn_tensor_load_to_lds(g0, g1, z4, z4, z8, 0);
}
#endif

// ======================= one-time prep kernels =======================
__global__ __launch_bounds__(256) void xcast_kernel(const float* __restrict__ x,
                                                    unsigned short* __restrict__ xb) {
    size_t i = ((size_t)blockIdx.x * 256 + threadIdx.x) * 2;
    unsigned pk = (unsigned)f32_to_bf16_rne(x[i]) |
                  ((unsigned)f32_to_bf16_rne(x[i + 1]) << 16);
    *(unsigned*)(xb + i) = pk;
}

// per-expert transpose + bf16 cast: src [E][R][C] fp32 -> dst [E][C][R] bf16
__global__ __launch_bounds__(256) void wtrans_kernel(const float* __restrict__ src,
                                                     unsigned short* __restrict__ dst,
                                                     int R, int C) {
    __shared__ unsigned short tile[32][33];
    const int e = blockIdx.z;
    const float* s = src + (size_t)e * R * C;
    unsigned short* d = dst + (size_t)e * R * C;
    const int c0 = blockIdx.x * 32, r0 = blockIdx.y * 32;
    const int tx = threadIdx.x & 31, ty = threadIdx.x >> 5;
#pragma unroll
    for (int i = 0; i < 32; i += 8)
        tile[ty + i][tx] = f32_to_bf16_rne(s[(size_t)(r0 + ty + i) * C + c0 + tx]);
    __syncthreads();
#pragma unroll
    for (int i = 0; i < 32; i += 8)
        d[(size_t)(c0 + ty + i) * R + r0 + tx] = tile[tx][ty + i];
}

// ======================= gating / routing kernel =======================
__global__ __launch_bounds__(128) void gate_kernel(
        const float* __restrict__ x, const float* __restrict__ Wg,
        int* __restrict__ counts, float* __restrict__ psum,
        int* __restrict__ idxl, float* __restrict__ wl) {
    __shared__ float sWg[DIM * NEXP];
    __shared__ float red[NEXP * 128];
    const int t = blockIdx.x, tid = threadIdx.x;

    for (int i = tid; i < DIM * NEXP; i += 128) sWg[i] = Wg[i];
    __syncthreads();

    float acc[NEXP];
#pragma unroll
    for (int e = 0; e < NEXP; ++e) acc[e] = 0.f;
    const float* xr = x + (size_t)t * DIM;
    for (int i = tid; i < DIM; i += 128) {
        float xv = xr[i];
#pragma unroll
        for (int e = 0; e < NEXP; ++e) acc[e] += xv * sWg[i * NEXP + e];
    }
#pragma unroll
    for (int e = 0; e < NEXP; ++e) red[e * 128 + tid] = acc[e];
    __syncthreads();
    for (int s = 64; s > 0; s >>= 1) {
        if (tid < s) {
#pragma unroll
            for (int e = 0; e < NEXP; ++e)
                red[e * 128 + tid] += red[e * 128 + tid + s];
        }
        __syncthreads();
    }
    if (tid == 0) {
        float lg[NEXP];
#pragma unroll
        for (int e = 0; e < NEXP; ++e) lg[e] = red[e * 128];
        float m = lg[0];
#pragma unroll
        for (int e = 1; e < NEXP; ++e) m = fmaxf(m, lg[e]);
        float p[NEXP], s = 0.f;
#pragma unroll
        for (int e = 0; e < NEXP; ++e) { p[e] = __expf(lg[e] - m); s += p[e]; }
        float inv = 1.f / s;
#pragma unroll
        for (int e = 0; e < NEXP; ++e) atomicAdd(&psum[e], p[e] * inv);
        int i0 = 0;
#pragma unroll
        for (int e = 1; e < NEXP; ++e) if (lg[e] > lg[i0]) i0 = e;
        int i1 = (i0 == 0) ? 1 : 0;
#pragma unroll
        for (int e = 0; e < NEXP; ++e) if (e != i0 && lg[e] > lg[i1]) i1 = e;
        float b = __expf(lg[i1] - lg[i0]);
        float w0 = 1.f / (1.f + b), w1 = b * w0;
        int s0 = atomicAdd(&counts[i0], 1);
        idxl[i0 * T_TOKENS + s0] = t;  wl[i0 * T_TOKENS + s0] = w0;
        int s1 = atomicAdd(&counts[i1], 1);
        idxl[i1 * T_TOKENS + s1] = t;  wl[i1 * T_TOKENS + s1] = w1;
    }
}

__global__ void loss_kernel(const float* __restrict__ psum,
                            float* __restrict__ out_loss) {
    if (threadIdx.x == 0 && blockIdx.x == 0) {
        float s = 0.f;
#pragma unroll
        for (int e = 0; e < NEXP; ++e) s += psum[e] * psum[e];
        *out_loss = (float)NEXP * s / (float)T_TOKENS;
    }
}

// ============== fused expert FFN (WMMA + pipelined TDM) ==============
// grid = (T_TOKENS/BM, NEXP), block = 512 threads = 16 wave32s.
// Weight tiles are double-buffered; wave 0 issues tile i+1 via TDM, then
// waits tensorcnt<=1 (tile i landed a full compute phase ago) -> the tensor
// DMA overlaps the WMMA work.
__global__ __launch_bounds__(512) void ffn_kernel(
        const unsigned short* __restrict__ xb,
        const unsigned short* __restrict__ W1Tb,   // [E][INNER][DIM] bf16
        const float* __restrict__ b1,
        const unsigned short* __restrict__ W2Tb,   // [E][DIM][INNER] bf16
        const float* __restrict__ b2,
        const int* __restrict__ counts, const int* __restrict__ idxl,
        const float* __restrict__ wl, float* __restrict__ out) {
    extern __shared__ __align__(16) unsigned char smem[];
    unsigned short* Xs   = (unsigned short*)smem;           // [64][LDX]
    unsigned short* W1T0 = Xs + BM * LDX;                   // [64][LD1] x2
    unsigned short* W1T1 = W1T0 + 64 * LD1;
    unsigned short* Hs   = W1T1 + 64 * LD1;                 // [64][LDH]
    unsigned short* W2T0 = Hs + BM * LDH;                   // [256][LD2] x2
    unsigned short* W2T1 = W2T0 + 256 * LD2;
    int*   Tok = (int*)(W2T1 + 256 * LD2);
    float* Wr  = (float*)(Tok + BM);

    const int e   = blockIdx.y;
    const int cnt = counts[e];
    const int m0  = blockIdx.x * BM;
    if (m0 >= cnt) return;

    const int tid  = threadIdx.x;
    const int lane = tid & 31;
    const int wave = tid >> 5;             // 0..15
    const int lrow = lane & 15;
    const int lhi  = lane >> 4;

    const unsigned short* W1e = W1Tb + (size_t)e * DIM * INNER;
    const unsigned short* W2e = W2Tb + (size_t)e * INNER * DIM;
    const float* b1e = b1 + e * INNER;
    const float* b2e = b2 + e * DIM;

    unsigned short* W1buf[2] = {W1T0, W1T1};
    unsigned short* W2buf[2] = {W2T0, W2T1};
#if USE_TDM
    const unsigned w1_off[2] = {(unsigned)(uintptr_t)(void*)W1T0,
                                (unsigned)(uintptr_t)(void*)W1T1};
    const unsigned w2_off[2] = {(unsigned)(uintptr_t)(void*)W2T0,
                                (unsigned)(uintptr_t)(void*)W2T1};
#endif

    if (tid < BM) {
        int valid = (m0 + tid) < cnt;
        Tok[tid] = valid ? idxl[e * T_TOKENS + m0 + tid] : 0;
        Wr[tid]  = valid ? wl[e * T_TOKENS + m0 + tid] : 0.f;
    }
    __syncthreads();

#if USE_TDM
    // pipeline prologue: start W1 slice 0 of chunk 0 while we gather x
    if (wave == 0)
        tdm_load_2d(w1_off[0], W1e, DIM, INNER, DIM, 128, 64, 5, 3);
#endif

    // gather x rows (already bf16) -> LDS, 8 halves per b128
    for (int g = tid; g < BM * (DIM / 8); g += 512) {
        int r = g >> 7, c8 = (g & 127) << 3;
        *(uint4*)(Xs + r * LDX + c8) =
            *(const uint4*)(xb + (size_t)Tok[r] * DIM + c8);
    }

    const v8f vzero = {0.f, 0.f, 0.f, 0.f, 0.f, 0.f, 0.f, 0.f};
    v8f acc[4][4];                         // [mf][ns]: col = ns*256 + wave*16
#pragma unroll
    for (int i = 0; i < 4; ++i)
#pragma unroll
        for (int j = 0; j < 4; ++j) acc[i][j] = vzero;

    const int mfh = wave >> 2, nfh = wave & 3;

    for (int chunk = 0; chunk < NCHUNK; ++chunk) {
        // ---- GEMM1: h[64][64] = X @ W1[:, chunk*64 .. +63] ----
        v8f hacc = vzero;
        for (int ks = 0; ks < 8; ++ks) {           // K slices of 128 over DIM
            __syncthreads();                        // prior readers of buffers done
#if USE_TDM
            if (wave == 0) {
                if (ks < 7) {                       // next W1 slice
                    tdm_load_2d(w1_off[(ks + 1) & 1],
                                W1e + (size_t)(chunk * 64) * DIM + (ks + 1) * 128,
                                DIM, INNER, DIM, 128, 64, 5, 3);
                } else {                            // first W2 slice of chunk
                    tdm_load_2d(w2_off[0],
                                W2e + (size_t)chunk * 64,
                                INNER, DIM, INNER, 64, 256, 4, 3);
                }
                __builtin_amdgcn_s_wait_tensorcnt((short)1);   // slice ks landed
            }
#else
            for (int g = tid; g < 64 * 16; g += 512) {
                int r = g >> 4, c8 = (g & 15) << 3;
                *(uint4*)(W1buf[ks & 1] + r * LD1 + c8) = *(const uint4*)(
                    W1e + (size_t)(chunk * 64 + r) * DIM + ks * 128 + c8);
            }
#endif
            __syncthreads();
            const unsigned short* W1cur = W1buf[ks & 1];
#pragma unroll
            for (int kk = 0; kk < 4; ++kk) {
                v16bf a = lds_load_a(Xs, LDX, mfh * 16 + lrow,
                                     ks * 128 + kk * 32 + (lhi << 4));
                v16bf b = lds_load_b(W1cur, LD1, nfh * 16 + lrow,
                                     kk * 32 + (lhi << 4));
                hacc = __builtin_amdgcn_wmma_f32_16x16x32_bf16(
                    false, a, false, b, (short)0, hacc, false, false);
            }
        }
        // bias + exact gelu, store own fragment of h (published by ns-loop barrier)
        {
            int   col  = nfh * 16 + lrow;
            float bias = b1e[chunk * 64 + col];
#pragma unroll
            for (int r = 0; r < 8; ++r) {
                int   row = mfh * 16 + r + (lhi << 3);
                float hv  = hacc[r] + bias;
                float gl  = 0.5f * hv * (1.0f + erff(hv * 0.70710678118654752f));
                Hs[row * LDH + col] = f32_to_bf16_rne(gl);
            }
        }
        // ---- GEMM2: out[64][1024] += h @ W2[chunk*64 .. +63, :] ----
#pragma unroll 1
        for (int ns = 0; ns < 4; ++ns) {           // N slices of 256
            __syncthreads();                        // publishes Hs; frees buffers
#if USE_TDM
            if (wave == 0) {
                const bool last = (chunk == NCHUNK - 1) && (ns == 3);
                if (!last) {
                    if (ns < 3) {                   // next W2 slice
                        tdm_load_2d(w2_off[(ns + 1) & 1],
                                    W2e + (size_t)((ns + 1) * 256) * INNER + chunk * 64,
                                    INNER, DIM, INNER, 64, 256, 4, 3);
                    } else {                        // W1 slice 0 of next chunk
                        tdm_load_2d(w1_off[0],
                                    W1e + (size_t)((chunk + 1) * 64) * DIM,
                                    DIM, INNER, DIM, 128, 64, 5, 3);
                    }
                    __builtin_amdgcn_s_wait_tensorcnt((short)1);  // slice ns landed
                } else {
                    __builtin_amdgcn_s_wait_tensorcnt((short)0);  // drain pipeline
                }
            }
#else
            for (int g = tid; g < 256 * 8; g += 512) {
                int r = g >> 3, c8 = (g & 7) << 3;
                *(uint4*)(W2buf[ns & 1] + r * LD2 + c8) = *(const uint4*)(
                    W2e + (size_t)(ns * 256 + r) * INNER + chunk * 64 + c8);
            }
#endif
            __syncthreads();
            const unsigned short* W2cur = W2buf[ns & 1];
#pragma unroll
            for (int mf = 0; mf < 4; ++mf) {
#pragma unroll
                for (int kk = 0; kk < 2; ++kk) {
                    v16bf a = lds_load_a(Hs, LDH, mf * 16 + lrow,
                                         kk * 32 + (lhi << 4));
                    v16bf b = lds_load_b(W2cur, LD2, wave * 16 + lrow,
                                         kk * 32 + (lhi << 4));
                    acc[mf][ns] = __builtin_amdgcn_wmma_f32_16x16x32_bf16(
                        false, a, false, b, (short)0, acc[mf][ns], false, false);
                }
            }
        }
        __syncthreads();    // last W2 readers done before Hs/W2 reuse next chunk
    }

    // weighted accumulate (each token gets exactly 2 commutative fp adds)
#pragma unroll
    for (int ns = 0; ns < 4; ++ns) {
        int   colg = ns * 256 + wave * 16 + lrow;
        float bias = b2e[colg];
#pragma unroll
        for (int mf = 0; mf < 4; ++mf) {
#pragma unroll
            for (int r = 0; r < 8; ++r) {
                int row = mf * 16 + r + (lhi << 3);
                if (m0 + row < cnt) {
                    atomicAdd(&out[(size_t)Tok[row] * DIM + colg],
                              Wr[row] * (acc[mf][ns][r] + bias));
                }
            }
        }
    }
}

// ============================ launch glue =============================
extern "C" void kernel_launch(void* const* d_in, const int* in_sizes, int n_in,
                              void* d_out, int out_size, void* d_ws,
                              size_t ws_size, hipStream_t stream) {
    (void)in_sizes; (void)n_in; (void)ws_size;
    const float* x  = (const float*)d_in[0];
    const float* Wg = (const float*)d_in[1];
    const float* W1 = (const float*)d_in[2];
    const float* b1 = (const float*)d_in[3];
    const float* W2 = (const float*)d_in[4];
    const float* b2 = (const float*)d_in[5];
    float* out = (float*)d_out;

    char* ws = (char*)d_ws;
    int*   counts = (int*)ws;                               // 8 ints
    float* psum   = (float*)(ws + 32);                      // 8 floats
    int*   idxl   = (int*)(ws + 64);                        // 8*8192 ints
    float* wl     = (float*)(ws + 64 + sizeof(int) * NEXP * T_TOKENS);
    size_t off    = 64 + (size_t)2 * sizeof(int) * NEXP * T_TOKENS;
    off = (off + 255) & ~(size_t)255;
    unsigned short* xb   = (unsigned short*)(ws + off);     // 16 MB bf16 x
    off += (size_t)T_TOKENS * DIM * 2;
    unsigned short* W1Tb = (unsigned short*)(ws + off);     // 64 MB bf16 W1^T
    off += (size_t)NEXP * DIM * INNER * 2;
    unsigned short* W2Tb = (unsigned short*)(ws + off);     // 64 MB bf16 W2^T

    (void)hipMemsetAsync(d_ws, 0, 64, stream);              // counts + psum
    (void)hipMemsetAsync(d_out, 0, sizeof(float) * (size_t)out_size, stream);

    // one-time precision/layout prep (stream-ordered, deterministic)
    xcast_kernel<<<(T_TOKENS * DIM) / 512, 256, 0, stream>>>(x, xb);
    {
        dim3 g1(INNER / 32, DIM / 32, NEXP);    // W1 [E][1024][4096] -> ^T
        wtrans_kernel<<<g1, 256, 0, stream>>>(W1, W1Tb, DIM, INNER);
        dim3 g2(DIM / 32, INNER / 32, NEXP);    // W2 [E][4096][1024] -> ^T
        wtrans_kernel<<<g2, 256, 0, stream>>>(W2, W2Tb, INNER, DIM);
    }

    gate_kernel<<<T_TOKENS, 128, 0, stream>>>(x, Wg, counts, psum, idxl, wl);
    loss_kernel<<<1, 32, 0, stream>>>(psum, out + (out_size - 1));

    dim3 grid(T_TOKENS / BM, NEXP);
    size_t shmem = (size_t)(BM * LDX + 2 * 64 * LD1 + BM * LDH + 2 * 256 * LD2) *
                       sizeof(unsigned short) +
                   BM * sizeof(int) + BM * sizeof(float);
    ffn_kernel<<<grid, 512, shmem, stream>>>(xb, W1Tb, b1, W2Tb, b2, counts,
                                             idxl, wl, out);
}
